// CustomSelfAttention_15917148799746
// MI455X (gfx1250) — matrix operand order repaired and checked by reading, not compiled, not run on hardware
//
#include <hip/hip_runtime.h>
#include <hip/hip_bf16.h>

#define N_TOK 8192
#define DMODEL 512
#define LN_EPS 1e-5f

typedef __attribute__((ext_vector_type(16))) __bf16        v16bf;
typedef __attribute__((ext_vector_type(8)))  float         v8f;
typedef __attribute__((ext_vector_type(4)))  unsigned int  u32x4;

union FragAB { u32x4 q[2]; v16bf v; };

__device__ __forceinline__ unsigned short f2bf(float f) {
  unsigned int u = __builtin_bit_cast(unsigned int, f);
  u += 0x7FFFu + ((u >> 16) & 1u);           // round-to-nearest-even
  return (unsigned short)(u >> 16);
}

// A-fragment (16x32 bf16, per ISA 7.12.2): lane row = lane%16,
// K-chunks: [k0 + (lane/16)*8, +8) and [k0 + 16 + (lane/16)*8, +8)
__device__ __forceinline__ v16bf load_a(const unsigned short* base, int ld,
                                        int row0, int k0, int lane) {
  int half = lane >> 4, l16 = lane & 15;
  const unsigned short* p = base + (row0 + l16) * ld + k0 + half * 8;
  FragAB f;
  f.q[0] = *(const u32x4*)(p);
  f.q[1] = *(const u32x4*)(p + 16);
  return f.v;
}

// B-fragment (32x16 bf16) read from "Bt" storage Bt[n][k] row-major:
// lane col n = lane%16, K-chunks: [k0 + (lane/16)*8, +8) and [k0 + 16 + (lane/16)*8, +8).
__device__ __forceinline__ v16bf load_b(const unsigned short* bt, int ld,
                                        int n0, int k0, int lane) {
  int half = lane >> 4, l16 = lane & 15;
  const unsigned short* p = bt + (n0 + l16) * ld + k0 + half * 8;
  FragAB f;
  f.q[0] = *(const u32x4*)(p);
  f.q[1] = *(const u32x4*)(p + 16);
  return f.v;
}

__device__ __forceinline__ v8f wmma_bf16(v16bf a, v16bf b, v8f c) {
  return __builtin_amdgcn_wmma_f32_16x16x32_bf16(
      /*neg_a=*/false, a, /*neg_b=*/false, b,
      /*c_mod=*/(short)0, c, /*reuse_a=*/false, /*reuse_b=*/false);
}

// ---------------------------------------------------------------- convert
__global__ void k_convert(const float* __restrict__ src,
                          unsigned short* __restrict__ dst, int n) {
  int i = blockIdx.x * blockDim.x + threadIdx.x;
  if (i < n) dst[i] = f2bf(src[i]);
}

// ---------------------------------------------------------------- QKV GEMM
// q = x@Wq^T + bq (row-major bf16), same for k; v stored transposed vt[D][N].
__global__ void k_qkv(const unsigned short* __restrict__ xb,
                      const unsigned short* __restrict__ Wqb,
                      const unsigned short* __restrict__ Wkb,
                      const unsigned short* __restrict__ Wvb,
                      const float* __restrict__ bq, const float* __restrict__ bk,
                      const float* __restrict__ bv,
                      unsigned short* __restrict__ qb,
                      unsigned short* __restrict__ kb,
                      unsigned short* __restrict__ vtb) {
  int lane = threadIdx.x & 31;
  int wave = threadIdx.x >> 5;
  int m0 = (blockIdx.x * 2 + wave) * 16;
  int n0 = blockIdx.y * 16;
  int mode = blockIdx.z;
  const unsigned short* W = (mode == 0) ? Wqb : (mode == 1) ? Wkb : Wvb;
  const float* bias = (mode == 0) ? bq : (mode == 1) ? bk : bv;

  v8f c = {};
#pragma unroll
  for (int k0 = 0; k0 < DMODEL; k0 += 32) {
    v16bf a = load_a(xb, DMODEL, m0, k0, lane);
    v16bf b = load_b(W, DMODEL, n0, k0, lane);   // W row-major == Bt layout
    c = wmma_bf16(a, b, c);
  }
  int half = lane >> 4, l16 = lane & 15;
  int col = n0 + l16;
  float bsv = bias[col];
#pragma unroll
  for (int r = 0; r < 8; ++r) {
    int row = m0 + r + half * 8;
    unsigned short val = f2bf(c[r] + bsv);
    if (mode == 0)      qb[row * DMODEL + col] = val;
    else if (mode == 1) kb[row * DMODEL + col] = val;
    else                vtb[col * N_TOK + row] = val;   // V transposed
  }
}

// ------------------------------------------------- pass A: softmax row stats
__global__ void k_stats(const unsigned short* __restrict__ qb,
                        const unsigned short* __restrict__ kb,
                        float* __restrict__ mrow, float* __restrict__ ilrow) {
  __shared__ unsigned short qs[4][16 * DMODEL];
  int lane = threadIdx.x & 31;
  int wave = threadIdx.x >> 5;
  int m0 = blockIdx.x * 64 + wave * 16;

  { // stage this wave's 16x512 Q tile into LDS (contiguous 16 KB)
    const u32x4* src = (const u32x4*)(qb + (size_t)m0 * DMODEL);
    u32x4* dst = (u32x4*)(&qs[wave][0]);
#pragma unroll
    for (int t = 0; t < 32; ++t) dst[t * 32 + lane] = src[t * 32 + lane];
  }

  float m[8], s[8];
#pragma unroll
  for (int r = 0; r < 8; ++r) { m[r] = -3.0e38f; s[r] = 0.f; }

  int l16 = lane & 15;
  for (int j0 = 0; j0 < N_TOK; j0 += 16) {
    if (j0 + 16 < N_TOK)
      __builtin_prefetch(kb + (size_t)(j0 + 16 + l16) * DMODEL, 0, 0);
    v8f c = {};
#pragma unroll
    for (int k0 = 0; k0 < DMODEL; k0 += 32) {
      v16bf a = load_a(&qs[wave][0], DMODEL, 0, k0, lane);
      v16bf b = load_b(kb, DMODEL, j0, k0, lane);  // K row-major == Bt
      c = wmma_bf16(a, b, c);
    }
#pragma unroll
    for (int r = 0; r < 8; ++r) {  // online softmax update per row segment
      float v = c[r];
      float nm = fmaxf(m[r], v);
      s[r] = s[r] * __expf(m[r] - nm) + __expf(v - nm);
      m[r] = nm;
    }
  }
  // combine across the 16 lanes holding the same row (xor stays in half-wave)
#pragma unroll
  for (int r = 0; r < 8; ++r) {
#pragma unroll
    for (int off = 1; off < 16; off <<= 1) {
      float om = __shfl_xor(m[r], off, 32);
      float os = __shfl_xor(s[r], off, 32);
      float nm = fmaxf(m[r], om);
      s[r] = s[r] * __expf(m[r] - nm) + os * __expf(om - nm);
      m[r] = nm;
    }
  }
  int half = lane >> 4;
  if (l16 == 0) {
#pragma unroll
    for (int r = 0; r < 8; ++r) {
      int row = m0 + r + half * 8;
      mrow[row] = m[r];
      ilrow[row] = 1.0f / s[r];
    }
  }
}

// ---------------------------- pass B: P = softmax(S), O = P@V, h = x + O
__global__ void k_attn_pv(const unsigned short* __restrict__ qb,
                          const unsigned short* __restrict__ kb,
                          const unsigned short* __restrict__ vtb,
                          const float* __restrict__ mrow,
                          const float* __restrict__ ilrow,
                          const float* __restrict__ x, float* __restrict__ H) {
  __shared__ unsigned short qs[16 * DMODEL];
  __shared__ unsigned short ps[2][16 * 32];
  int lane = threadIdx.x & 31;
  int wave = threadIdx.x >> 5;
  int m0 = blockIdx.x * 16;
  int dbase = wave * 256;                 // each wave owns half of D

  { // cooperative 16 KB Q-tile stage
    const u32x4* src = (const u32x4*)(qb + (size_t)m0 * DMODEL);
    u32x4* dst = (u32x4*)qs;
    int tid = threadIdx.x;
#pragma unroll
    for (int t = 0; t < 16; ++t) dst[t * 64 + tid] = src[t * 64 + tid];
  }
  __syncthreads();

  int half = lane >> 4, l16 = lane & 15;
  float mr[8], il[8];
#pragma unroll
  for (int r = 0; r < 8; ++r) {
    int row = m0 + r + half * 8;
    mr[r] = mrow[row];
    il[r] = ilrow[row];
  }
  v8f zero = {};
  v8f o[16];
#pragma unroll
  for (int dt = 0; dt < 16; ++dt) o[dt] = zero;

  for (int j0 = 0; j0 < N_TOK; j0 += 32) {
#pragma unroll
    for (int nt = 0; nt < 2; ++nt) {
      v8f c = {};
#pragma unroll
      for (int k0 = 0; k0 < DMODEL; k0 += 32) {
        v16bf a = load_a(qs, DMODEL, 0, k0, lane);
        v16bf b = load_b(kb, DMODEL, j0 + nt * 16, k0, lane);
        c = wmma_bf16(a, b, c);
      }
      // normalized P in bf16, re-layout C->A via per-wave LDS buffer
#pragma unroll
      for (int r = 0; r < 8; ++r) {
        float p = __expf(c[r] - mr[r]) * il[r];
        ps[wave][(r + half * 8) * 32 + nt * 16 + l16] = f2bf(p);
      }
    }
    v16bf pa = load_a(&ps[wave][0], 32, 0, 0, lane);
#pragma unroll
    for (int dt = 0; dt < 16; ++dt) {
      v16bf b = load_b(vtb, N_TOK, dbase + dt * 16, j0, lane); // V^T == Bt
      o[dt] = wmma_bf16(pa, b, o[dt]);
    }
  }
  // residual: h = x + attn_out (f32)
#pragma unroll
  for (int dt = 0; dt < 16; ++dt) {
    int col = dbase + dt * 16 + l16;
#pragma unroll
    for (int r = 0; r < 8; ++r) {
      int row = m0 + r + half * 8;
      H[(size_t)row * DMODEL + col] = x[(size_t)row * DMODEL + col] + o[dt][r];
    }
  }
}

// ------------------------------------------------- LayerNorm (in place in H)
__global__ void k_ln(float* __restrict__ H, const float* __restrict__ gamma,
                     const float* __restrict__ beta) {
  int lane = threadIdx.x & 31;
  int wave = threadIdx.x >> 5;
  int row = blockIdx.x * 8 + wave;
  float* h = H + (size_t)row * DMODEL;
  float v[16];
  float sum = 0.f;
#pragma unroll
  for (int i = 0; i < 16; ++i) { v[i] = h[lane + i * 32]; sum += v[i]; }
#pragma unroll
  for (int off = 16; off; off >>= 1) sum += __shfl_xor(sum, off, 32);
  float mu = sum * (1.0f / DMODEL);
  float var = 0.f;
#pragma unroll
  for (int i = 0; i < 16; ++i) { float d = v[i] - mu; var += d * d; }
#pragma unroll
  for (int off = 16; off; off >>= 1) var += __shfl_xor(var, off, 32);
  var *= (1.0f / DMODEL);
  float rs = rsqrtf(var + LN_EPS);
#pragma unroll
  for (int i = 0; i < 16; ++i) {
    int col = lane + i * 32;
    h[col] = (v[i] - mu) * rs * gamma[col] + beta[col];
  }
}

// --------------------------------- deterministic mean pool over rows -> out
__global__ void k_pool(const float* __restrict__ R, float* __restrict__ out) {
  int col = blockIdx.x * blockDim.x + threadIdx.x;
  if (col >= DMODEL) return;
  float s = 0.f;
  for (int row = 0; row < N_TOK; ++row) s += R[(size_t)row * DMODEL + col];
  out[col] = s * (1.0f / N_TOK);
}

extern "C" void kernel_launch(void* const* d_in, const int* in_sizes, int n_in,
                              void* d_out, int out_size, void* d_ws, size_t ws_size,
                              hipStream_t stream) {
  const float* x     = (const float*)d_in[0];
  const float* Wq    = (const float*)d_in[1];
  const float* bq    = (const float*)d_in[2];
  const float* Wk    = (const float*)d_in[3];
  const float* bk    = (const float*)d_in[4];
  const float* Wv    = (const float*)d_in[5];
  const float* bv    = (const float*)d_in[6];
  const float* gamma = (const float*)d_in[7];
  const float* beta  = (const float*)d_in[8];
  float* out = (float*)d_out;

  char* ws = (char*)d_ws;
  size_t off = 0;
  auto alloc = [&](size_t bytes) -> void* {
    void* p = ws + off;
    off += (bytes + 255) & ~(size_t)255;
    return p;
  };
  unsigned short* xb   = (unsigned short*)alloc((size_t)N_TOK * DMODEL * 2);
  unsigned short* Wqb  = (unsigned short*)alloc((size_t)DMODEL * DMODEL * 2);
  unsigned short* Wkb  = (unsigned short*)alloc((size_t)DMODEL * DMODEL * 2);
  unsigned short* Wvb  = (unsigned short*)alloc((size_t)DMODEL * DMODEL * 2);
  unsigned short* qb   = (unsigned short*)alloc((size_t)N_TOK * DMODEL * 2);
  unsigned short* kb   = (unsigned short*)alloc((size_t)N_TOK * DMODEL * 2);
  unsigned short* vtb  = (unsigned short*)alloc((size_t)DMODEL * N_TOK * 2);
  float*          mrow = (float*)alloc((size_t)N_TOK * 4);
  float*          ilrw = (float*)alloc((size_t)N_TOK * 4);
  float*          H    = (float*)alloc((size_t)N_TOK * DMODEL * 4);

  int nx = N_TOK * DMODEL;
  int nw = DMODEL * DMODEL;
  k_convert<<<(nx + 255) / 256, 256, 0, stream>>>(x, xb, nx);
  k_convert<<<(nw + 255) / 256, 256, 0, stream>>>(Wq, Wqb, nw);
  k_convert<<<(nw + 255) / 256, 256, 0, stream>>>(Wk, Wkb, nw);
  k_convert<<<(nw + 255) / 256, 256, 0, stream>>>(Wv, Wvb, nw);

  k_qkv<<<dim3(N_TOK / 32, DMODEL / 16, 3), 64, 0, stream>>>(
      xb, Wqb, Wkb, Wvb, bq, bk, bv, qb, kb, vtb);

  k_stats<<<N_TOK / 64, 128, 0, stream>>>(qb, kb, mrow, ilrw);

  k_attn_pv<<<N_TOK / 16, 64, 0, stream>>>(qb, kb, vtb, mrow, ilrw, x, H);

  k_ln<<<N_TOK / 8, 256, 0, stream>>>(H, gamma, beta);
  k_pool<<<2, 256, 0, stream>>>(H, out);
}
